// N3RModelFast4GB_14680198218397
// MI455X (gfx1250) — matrix operand order repaired and checked by reading, not compiled, not run on hardware
//
#include <hip/hip_runtime.h>

typedef _Float16 v16h __attribute__((ext_vector_type(16)));
typedef _Float16 v8h  __attribute__((ext_vector_type(8)));
typedef _Float16 v4h  __attribute__((ext_vector_type(4)));
typedef float    v8f  __attribute__((ext_vector_type(8)));

#define WPB 8  // waves per block (256 threads, wave32)

__global__ __launch_bounds__(256, 1) void nerf_mlp_wmma(
    const _Float16* __restrict__ coords,
    const _Float16* __restrict__ W1, const _Float16* __restrict__ b1,
    const _Float16* __restrict__ W2, const _Float16* __restrict__ b2,
    const _Float16* __restrict__ W3, const _Float16* __restrict__ b3,
    _Float16* __restrict__ out, int npts)
{
    // Per-wave 16x64 fp16 staging buffer for the C/D -> A-layout transpose.
    __shared__ alignas(16) _Float16 hbuf[WPB][16][64];

    const int tid  = threadIdx.x;
    const int wid  = tid >> 5;
    const int lane = tid & 31;
    const int col  = lane & 15;   // N index (B/C/D) or M index (A)
    const int half = lane >> 4;   // K-half selector

    // ---------------- Preload weights into WMMA B-operand layout ----------------
    // B layout (16-bit, 32x16): lane<16 -> column N=lane, elems = K{0..7,16..23};
    //                           lane>=16 -> column N=lane-16, elems = K{8..15,24..31}.
    v16h w1b[4], w2b[2][4], w3b[2];
    #pragma unroll
    for (int j = 0; j < 4; ++j) {
        #pragma unroll
        for (int e = 0; e < 16; ++e) {
            const int k = 8 * half + e + ((e >= 8) ? 8 : 0);  // padded K (27..31 -> 0)
            w1b[j][e] = (k < 27) ? W1[k * 64 + 16 * j + col] : (_Float16)0.0f;
        }
    }
    #pragma unroll
    for (int kc = 0; kc < 2; ++kc)
        #pragma unroll
        for (int j = 0; j < 4; ++j)
            #pragma unroll
            for (int e = 0; e < 16; ++e) {
                const int k = 32 * kc + 8 * half + e + ((e >= 8) ? 8 : 0);
                w2b[kc][j][e] = W2[k * 64 + 16 * j + col];
            }
    #pragma unroll
    for (int kc = 0; kc < 2; ++kc)
        #pragma unroll
        for (int e = 0; e < 16; ++e) {
            const int k = 32 * kc + 8 * half + e + ((e >= 8) ? 8 : 0);
            w3b[kc][e] = (col < 4) ? W3[k * 4 + col] : (_Float16)0.0f;
        }

    // Bias splats (C/D tiles: N depends only on lane, so one float per lane/tile).
    float bias1[4], bias2[4];
    #pragma unroll
    for (int j = 0; j < 4; ++j) {
        bias1[j] = (float)b1[16 * j + col];
        bias2[j] = (float)b2[16 * j + col];
    }
    const float bias3 = (col < 4) ? (float)b3[col] : 0.0f;

    const int numTiles   = (npts + 15) >> 4;
    const int totalWaves = gridDim.x * WPB;
    const int wgid       = blockIdx.x * WPB + wid;

    for (int t = wgid; t < numTiles; t += totalWaves) {
        const int base = t << 4;
        const int p    = base + col;
        const int pc   = (p < npts) ? p : 0;

        // Prefetch next tile's coords (global_prefetch_b8).
        const int tn = t + totalWaves;
        if (tn < numTiles)
            __builtin_prefetch(&coords[(size_t)(tn << 4) * 3], 0, 1);

        const float x = (float)coords[3 * pc + 0];
        const float y = (float)coords[3 * pc + 1];
        const float z = (float)coords[3 * pc + 2];

        // Positional encoding: sin/cos(2^l * c), torch order
        // enc = [x,y,z, sin0(xyz), cos0(xyz), sin1.., cos1.., sin2.., cos2.., sin3.., cos3..]
        float sv[4][3], cv[4][3];
        #pragma unroll
        for (int l = 0; l < 4; ++l) {
            const float f  = (float)(1 << l);
            const float a0 = x * f, a1 = y * f, a2 = z * f;
            sv[l][0] = __sinf(a0); sv[l][1] = __sinf(a1); sv[l][2] = __sinf(a2);
            cv[l][0] = __cosf(a0); cv[l][1] = __cosf(a1); cv[l][2] = __cosf(a2);
        }
        auto encf = [&](int k) -> float {
            if (k == 0) return x;
            if (k == 1) return y;
            if (k == 2) return z;
            if (k >= 27) return 0.0f;          // zero padding K=27..31
            const int l = (k - 3) / 6, r = (k - 3) % 6;
            return (r < 3) ? sv[l][r] : cv[l][r - 3];
        };

        // A-operand (16x32 fp16): lane<16 elems = K{0..7,16..23}; lane>=16 = K{8..15,24..31}
        v16h aEnc;
        #pragma unroll
        for (int i = 0; i < 8; ++i) {
            const float lo0 = encf(i),      hi0 = encf(8 + i);
            const float lo1 = encf(16 + i), hi1 = encf(24 + i);
            aEnc[i]     = (_Float16)(half ? hi0 : lo0);
            aEnc[8 + i] = (_Float16)(half ? hi1 : lo1);
        }

        // ---------------- Layer 1: enc(16x32) @ W1(32x64) + b1 ----------------
        v8f c1[4];
        #pragma unroll
        for (int j = 0; j < 4; ++j) {
            v8f c;
            #pragma unroll
            for (int r = 0; r < 8; ++r) c[r] = bias1[j];
            c1[j] = __builtin_amdgcn_wmma_f32_16x16x32_f16(
                false, aEnc, false, w1b[j], (short)0, c, false, false);
        }

        // fp16 convert (pairs -> v_cvt_pk_f16_f32), packed ReLU (v_pk_max_num_f16),
        // C/D layout -> hbuf[M][N]; then reload in A layout.
        __builtin_amdgcn_wave_barrier();
        #pragma unroll
        for (int j = 0; j < 4; ++j) {
            v8h hh;
            #pragma unroll
            for (int r = 0; r < 8; ++r) hh[r] = (_Float16)c1[j][r];
            const v8h zero = {};
            hh = __builtin_elementwise_max(hh, zero);   // packed f16 ReLU
            #pragma unroll
            for (int r = 0; r < 8; ++r)
                hbuf[wid][r + 8 * half][16 * j + col] = hh[r];
        }
        __builtin_amdgcn_wave_barrier();

        v16h a2[2];
        #pragma unroll
        for (int kc = 0; kc < 2; ++kc) {
            const v8h lo = *(const v8h*)&hbuf[wid][col][32 * kc + 8 * half];
            const v8h hi = *(const v8h*)&hbuf[wid][col][32 * kc + 16 + 8 * half];
            a2[kc] = __builtin_shufflevector(lo, hi,
                0, 1, 2, 3, 4, 5, 6, 7, 8, 9, 10, 11, 12, 13, 14, 15);
        }

        // ---------------- Layer 2: h1(16x64) @ W2(64x64) + b2 ----------------
        v8f c2[4];
        #pragma unroll
        for (int j = 0; j < 4; ++j) {
            v8f c;
            #pragma unroll
            for (int r = 0; r < 8; ++r) c[r] = bias2[j];
            c = __builtin_amdgcn_wmma_f32_16x16x32_f16(
                false, a2[0], false, w2b[0][j], (short)0, c, false, false);
            c2[j] = __builtin_amdgcn_wmma_f32_16x16x32_f16(
                false, a2[1], false, w2b[1][j], (short)0, c, false, false);
        }

        __builtin_amdgcn_wave_barrier();
        #pragma unroll
        for (int j = 0; j < 4; ++j) {
            v8h hh;
            #pragma unroll
            for (int r = 0; r < 8; ++r) hh[r] = (_Float16)c2[j][r];
            const v8h zero = {};
            hh = __builtin_elementwise_max(hh, zero);   // packed f16 ReLU
            #pragma unroll
            for (int r = 0; r < 8; ++r)
                hbuf[wid][r + 8 * half][16 * j + col] = hh[r];
        }
        __builtin_amdgcn_wave_barrier();

        v16h a3[2];
        #pragma unroll
        for (int kc = 0; kc < 2; ++kc) {
            const v8h lo = *(const v8h*)&hbuf[wid][col][32 * kc + 8 * half];
            const v8h hi = *(const v8h*)&hbuf[wid][col][32 * kc + 16 + 8 * half];
            a3[kc] = __builtin_shufflevector(lo, hi,
                0, 1, 2, 3, 4, 5, 6, 7, 8, 9, 10, 11, 12, 13, 14, 15);
        }

        // ---------------- Layer 3: h2(16x64) @ W3(64x16 padded) + b3 ----------------
        v8f d;
        #pragma unroll
        for (int r = 0; r < 8; ++r) d[r] = bias3;
        d = __builtin_amdgcn_wmma_f32_16x16x32_f16(
            false, a3[0], false, w3b[0], (short)0, d, false, false);
        d = __builtin_amdgcn_wmma_f32_16x16x32_f16(
            false, a3[1], false, w3b[1], (short)0, d, false, false);

        // ---------------- Output: D cols 0..3 -> LDS -> coalesced b64 stores ----------------
        __builtin_amdgcn_wave_barrier();
        if (col < 4) {
            v8h oh;
            #pragma unroll
            for (int r = 0; r < 8; ++r) oh[r] = (_Float16)d[r];
            #pragma unroll
            for (int r = 0; r < 8; ++r)
                hbuf[wid][r + 8 * half][col] = oh[r];
        }
        __builtin_amdgcn_wave_barrier();
        if (half == 0) {
            const int pr = base + col;
            if (pr < npts) {
                const v4h o = *(const v4h*)&hbuf[wid][col][0];
                *(v4h*)&out[4 * (size_t)pr] = o;
            }
        }
        __builtin_amdgcn_wave_barrier();  // protect LDS WAR vs next iteration
    }
}

extern "C" void kernel_launch(void* const* d_in, const int* in_sizes, int n_in,
                              void* d_out, int out_size, void* d_ws, size_t ws_size,
                              hipStream_t stream) {
    (void)n_in; (void)d_ws; (void)ws_size; (void)out_size;
    const _Float16* coords = (const _Float16*)d_in[0];
    const _Float16* W1     = (const _Float16*)d_in[1];
    const _Float16* b1     = (const _Float16*)d_in[2];
    const _Float16* W2     = (const _Float16*)d_in[3];
    const _Float16* b2     = (const _Float16*)d_in[4];
    const _Float16* W3     = (const _Float16*)d_in[5];
    const _Float16* b3     = (const _Float16*)d_in[6];
    _Float16* out          = (_Float16*)d_out;

    const int npts = in_sizes[0] / 3;
    if (npts <= 0) return;

    const int numTiles = (npts + 15) >> 4;
    int blocks = (numTiles + WPB - 1) / WPB;
    if (blocks > 1024) blocks = 1024;
    if (blocks < 1) blocks = 1;

    nerf_mlp_wmma<<<blocks, 256, 0, stream>>>(coords, W1, b1, W2, b2, W3, b3,
                                              out, npts);
}
